// Hla2pepAttentionFusion_54528904790552
// MI455X (gfx1250) — matrix operand order repaired and checked by reading, not compiled, not run on hardware
//
#include <hip/hip_runtime.h>
#include <math.h>

typedef __attribute__((ext_vector_type(16))) __bf16 v16bf;
typedef __attribute__((ext_vector_type(8)))  __bf16 v8bf;
typedef __attribute__((ext_vector_type(8)))  float  v8f;

#define WMMA_BF16(a, b, c) \
  __builtin_amdgcn_wmma_f32_16x16x32_bf16(false, (a), false, (b), (short)0, (c), false, false)

#define D_MODEL 64
#define HIDDEN  64
#define LH      128
#define LP      32
#define NH      4
#define HD      16

__device__ __forceinline__ __bf16 f2bf(float f) { return (__bf16)f; }
__device__ __forceinline__ float bf2f(__bf16 h) { return (float)h; }

// A fragment: 16x32 bf16 tile, p points at (row0, k0), leading dim ld (elements).
// lanes 0-15 hold K 0-15 (e0..15); lanes 16-31 hold K 16-31.
__device__ __forceinline__ v16bf ldsA(const __bf16* p, int ld, int lane) {
  const __bf16* q = p + (lane & 15) * ld + ((lane >> 4) << 4);
  v8bf lo = *(const v8bf*)(q);
  v8bf hi = *(const v8bf*)(q + 8);
  v16bf a;
#pragma unroll
  for (int e = 0; e < 8; ++e) { a[e] = lo[e]; a[8 + e] = hi[e]; }
  return a;
}

// B fragment from LDS staged in fragment order (512 elems/frag, lane-contiguous).
__device__ __forceinline__ v16bf ldsB(const __bf16* base, int f, int lane) {
  return *(const v16bf*)(base + (((f << 5) + lane) << 4));
}

// Stage a KxN f32 weight (row-major) into LDS in B-fragment order.
__device__ __forceinline__ void stage_W_frag(const float* __restrict__ Wg, __bf16* dst,
                                             int K, int N, int tid) {
  const int NT = N >> 4;
  const int nslot = (K * N) >> 4;
  for (int s = tid; s < nslot; s += 256) {
    const int f  = s >> 5, l = s & 31;
    const int kt = f / NT, nt = f - kt * NT;
    const int k0 = (kt << 5) + ((l >> 4) << 4);
    const int n  = (nt << 4) + (l & 15);
    const float* gp = Wg + k0 * N + n;
    v16bf tmp;
#pragma unroll
    for (int e = 0; e < 16; ++e) tmp[e] = f2bf(gp[e * N]);
    *(v16bf*)(dst + (s << 4)) = tmp;
  }
}

// Stage contiguous f32 -> bf16 row-major into LDS, vectorized.
__device__ __forceinline__ void stage_act(const float* __restrict__ src, __bf16* dst,
                                          int count, int tid) {
  for (int i = tid * 8; i < count; i += 256 * 8) {
    float4 f0 = *(const float4*)(src + i);
    float4 f1 = *(const float4*)(src + i + 4);
    v8bf t;
    t[0] = f2bf(f0.x); t[1] = f2bf(f0.y); t[2] = f2bf(f0.z); t[3] = f2bf(f0.w);
    t[4] = f2bf(f1.x); t[5] = f2bf(f1.y); t[6] = f2bf(f1.z); t[7] = f2bf(f1.w);
    *(v8bf*)(dst + i) = t;
  }
}

__global__ __launch_bounds__(256, 1)
void hla2pep_fused(const float* __restrict__ hlain, const float* __restrict__ pepin,
                   const float* __restrict__ Whla,  const float* __restrict__ bhla,
                   const float* __restrict__ Wpep,  const float* __restrict__ bpep,
                   const float* __restrict__ relb,
                   const float* __restrict__ lng,   const float* __restrict__ lnbv,
                   const float* __restrict__ W1g,   const float* __restrict__ b1g,
                   const float* __restrict__ W2g,   const float* __restrict__ b2g,
                   const float* __restrict__ Wog,   const float* __restrict__ bog,
                   float* __restrict__ out) {
  const int bidx = blockIdx.x;
  const int tid  = threadIdx.x;
  const int wave = tid >> 5;
  const int lane = tid & 31;
  const int ln15 = lane & 15;
  const int lhi  = lane >> 4;
  const int mb   = (wave << 4) + (lhi << 3);   // row base for this lane's D fragment

  // LDS carve: 100KB total, regions reused across pipeline phases.
  __shared__ __align__(64) unsigned char smem[102400];
  __bf16* sIn   = (__bf16*)(smem);          // 16KB: hla input bf16 [128][64]
  float*  sSc   = (float*)(smem);           //       -> scores f32 [128][32]
  __bf16* sH    = (__bf16*)(smem);          //       -> LN out bf16 [128][64]
  __bf16* sH2   = (__bf16*)(smem);          //       -> pre-outproj bf16 [128][64]
  __bf16* sQ    = (__bf16*)(smem + 16384);  // 16KB: hla proj = q = residual [128][64]
  float*  sCtx  = (float*)(smem + 32768);   // 32KB: ctx f32 [128][64]
  __bf16* sH1   = (__bf16*)(smem + 32768);  //       -> FFN hidden bf16 [128][128]
  __bf16* sPepB = (__bf16*)(smem + 65536);  //  4KB: k/v, ctx-B fragment layout (4 frags)
  __bf16* sAux  = (__bf16*)(smem + 69632);  //  8KB: pep input staging -> attn bf16 [128][32]
  __bf16* sW    = (__bf16*)(smem + 77824);  // 16KB: weights, B-fragment layout
  __bf16* sScB  = (__bf16*)(smem + 94208);  //  8KB: k^T, scores-B fragment layout (8 frags)

  const __bf16 zbf = f2bf(0.f);

  // ---- Phase 0: stage inputs + first two weights; zero-pad scores-B region --
  stage_act(hlain + (size_t)bidx * (LH * D_MODEL), sIn,  LH * D_MODEL, tid);
  stage_act(pepin + (size_t)bidx * (LP * D_MODEL), sAux, LP * D_MODEL, tid);
  stage_W_frag(Whla, sW,        64, 64, tid);
  stage_W_frag(Wpep, sW + 4096, 64, 64, tid);
  {
    v16bf z = {};
    *(v16bf*)(sScB + (tid << 4)) = z;       // 256 slots, one per thread
  }
  __syncthreads();

  // ---- Phase 1: hla projection (q / residual), M=128 N=64 K=64 -------------
  {
    v16bf a0 = ldsA(sIn + (wave << 4) * 64 + 0,  64, lane);
    v16bf a1 = ldsA(sIn + (wave << 4) * 64 + 32, 64, lane);
    __bf16* p = sQ + mb * 64 + ln15;
#pragma unroll
    for (int nt = 0; nt < 4; ++nt) {
      v8f acc = {};
      acc = WMMA_BF16(a0, ldsB(sW, 0 * 4 + nt, lane), acc);
      acc = WMMA_BF16(a1, ldsB(sW, 1 * 4 + nt, lane), acc);
      const float bias = bhla[nt * 16 + ln15];
#pragma unroll
      for (int v = 0; v < 8; ++v)
        p[v * 64 + nt * 16] = f2bf(acc[v] + bias);
    }
  }
  // ---- pep projection (k/v), M=32 N=64 K=64 (one tile per wave) ------------
  // h == nt, d == ln15; ctx-B scatter is contiguous in v -> one b128 store.
  {
    const int mt = wave >> 2, nt = wave & 3;
    v16bf a0 = ldsA(sAux + (mt << 4) * 64 + 0,  64, lane);
    v16bf a1 = ldsA(sAux + (mt << 4) * 64 + 32, 64, lane);
    v8f acc = {};
    acc = WMMA_BF16(a0, ldsB(sW + 4096, 0 * 4 + nt, lane), acc);
    acc = WMMA_BF16(a1, ldsB(sW + 4096, 1 * 4 + nt, lane), acc);
    const float bias = bpep[nt * 16 + ln15];
    v8bf pk;
#pragma unroll
    for (int v = 0; v < 8; ++v) pk[v] = f2bf(acc[v] + bias);
    // ctx B frag: frag h=nt; lane' = ln15 + 16*mt; elems (lhi*8+v) contiguous
    *(v8bf*)(sPepB + (nt << 9) + ((ln15 + (mt << 4)) << 4) + (lhi << 3)) = pk;
    // scores B frag: frag nt*2+mt; lane'' = lhi*8+v (stride 16); e = ln15
    __bf16* ps = sScB + (((nt << 1) + mt) << 9) + (lhi << 7) + ln15;
#pragma unroll
    for (int v = 0; v < 8; ++v) ps[v << 4] = pk[v];
  }
  __syncthreads();

  // ---- Phase 2: attention, head by head ------------------------------------
  for (int h = 0; h < NH; ++h) {
    // scores = q_h @ k_h^T * 0.25 + rel_bias   (K=16, zero-padded to 32)
    {
      v16bf a;
      {
        const __bf16* p = sQ + ((wave << 4) + ln15) * 64 + h * HD;
        v8bf lo = *(const v8bf*)(p);
        v8bf hi = *(const v8bf*)(p + 8);
#pragma unroll
        for (int e = 0; e < 8; ++e) { a[e] = lhi ? zbf : lo[e]; a[8 + e] = lhi ? zbf : hi[e]; }
      }
      const float* rb = relb + h * 16384 + mb * 128 + ln15;
      float* psc = sSc + mb * 32 + ln15;
#pragma unroll
      for (int nt = 0; nt < 2; ++nt) {
        v8f acc = {};
        acc = WMMA_BF16(a, ldsB(sScB, (h << 1) + nt, lane), acc);
#pragma unroll
        for (int v = 0; v < 8; ++v)
          psc[v * 32 + nt * 16] = acc[v] * 0.25f + rb[v * 128 + nt * 16];
      }
    }
    __syncthreads();
    // row softmax over 32 cols (one row per thread), vectorized LDS access
    if (tid < 128) {
      float x[32];
      const float4* p4 = (const float4*)(sSc + tid * 32);
#pragma unroll
      for (int i = 0; i < 8; ++i) {
        float4 t = p4[i];
        x[4 * i] = t.x; x[4 * i + 1] = t.y; x[4 * i + 2] = t.z; x[4 * i + 3] = t.w;
      }
      float mx = x[0];
#pragma unroll
      for (int c = 1; c < 32; ++c) mx = fmaxf(mx, x[c]);
      float s = 0.f;
#pragma unroll
      for (int c = 0; c < 32; ++c) { x[c] = __expf(x[c] - mx); s += x[c]; }
      const float inv = 1.f / s;
#pragma unroll
      for (int j = 0; j < 4; ++j) {
        v8bf o;
#pragma unroll
        for (int e = 0; e < 8; ++e) o[e] = f2bf(x[8 * j + e] * inv);
        *(v8bf*)(sAux + tid * 32 + 8 * j) = o;
      }
    }
    __syncthreads();
    // ctx_h = attn @ v_h : M=128 N=16 K=32 (one tile per wave)
    {
      v16bf a = ldsA(sAux + (wave << 4) * 32, 32, lane);
      v8f acc = {};
      acc = WMMA_BF16(a, ldsB(sPepB, h, lane), acc);
      float* pc = sCtx + mb * 64 + h * HD + ln15;
#pragma unroll
      for (int v = 0; v < 8; ++v) pc[v * 64] = acc[v];
    }
    __syncthreads();
  }

  // ---- Phase 3: LayerNorm (row per thread, vectorized) + stage W1 ----------
  if (tid < 128) {
    float x[64];
    const float4* p4 = (const float4*)(sCtx + tid * 64);
#pragma unroll
    for (int i = 0; i < 16; ++i) {
      float4 t = p4[i];
      x[4 * i] = t.x; x[4 * i + 1] = t.y; x[4 * i + 2] = t.z; x[4 * i + 3] = t.w;
    }
    float mu = 0.f;
#pragma unroll
    for (int c = 0; c < 64; ++c) mu += x[c];
    mu *= (1.f / 64.f);
    float var = 0.f;
#pragma unroll
    for (int c = 0; c < 64; ++c) { float d = x[c] - mu; var += d * d; }
    var *= (1.f / 64.f);
    const float rs = rsqrtf(var + 1e-5f);
#pragma unroll
    for (int j = 0; j < 8; ++j) {
      float4 g0 = *(const float4*)(lng + 8 * j);
      float4 g1 = *(const float4*)(lng + 8 * j + 4);
      float4 c0 = *(const float4*)(lnbv + 8 * j);
      float4 c1 = *(const float4*)(lnbv + 8 * j + 4);
      const float gg[8] = {g0.x, g0.y, g0.z, g0.w, g1.x, g1.y, g1.z, g1.w};
      const float bb[8] = {c0.x, c0.y, c0.z, c0.w, c1.x, c1.y, c1.z, c1.w};
      v8bf o;
#pragma unroll
      for (int e = 0; e < 8; ++e)
        o[e] = f2bf((x[8 * j + e] - mu) * rs * gg[e] + bb[e]);
      *(v8bf*)(sH + tid * 64 + 8 * j) = o;
    }
  }
  stage_W_frag(W1g, sW, 64, 128, tid);
  __syncthreads();

  // ---- Phase 4: FFN1 + exact GELU, M=128 N=128 K=64 ------------------------
  {
    v16bf a0 = ldsA(sH + (wave << 4) * 64 + 0,  64, lane);
    v16bf a1 = ldsA(sH + (wave << 4) * 64 + 32, 64, lane);
    __bf16* p = sH1 + mb * 128 + ln15;
#pragma unroll
    for (int nt = 0; nt < 8; ++nt) {
      v8f acc = {};
      acc = WMMA_BF16(a0, ldsB(sW, 0 * 8 + nt, lane), acc);
      acc = WMMA_BF16(a1, ldsB(sW, 1 * 8 + nt, lane), acc);
      const float bias = b1g[nt * 16 + ln15];
#pragma unroll
      for (int v = 0; v < 8; ++v) {
        float xx = acc[v] + bias;
        float g = 0.5f * xx * (1.f + erff(xx * 0.70710678118f));
        p[v * 128 + nt * 16] = f2bf(g);
      }
    }
  }
  __syncthreads();
  stage_W_frag(W2g, sW, 128, 64, tid);
  __syncthreads();

  // ---- Phase 5: FFN2 + residual, M=128 N=64 K=128 --------------------------
  {
    v16bf aF[4];
#pragma unroll
    for (int kt = 0; kt < 4; ++kt)
      aF[kt] = ldsA(sH1 + (wave << 4) * 128 + kt * 32, 128, lane);
    __bf16* p = sH2 + mb * 64 + ln15;
    const __bf16* q = sQ + mb * 64 + ln15;
#pragma unroll
    for (int nt = 0; nt < 4; ++nt) {
      v8f acc = {};
#pragma unroll
      for (int kt = 0; kt < 4; ++kt)
        acc = WMMA_BF16(aF[kt], ldsB(sW, kt * 4 + nt, lane), acc);
      const float bias = b2g[nt * 16 + ln15];
#pragma unroll
      for (int v = 0; v < 8; ++v)
        p[v * 64 + nt * 16] = f2bf(acc[v] + bias + bf2f(q[v * 64 + nt * 16]));
    }
  }
  __syncthreads();
  stage_W_frag(Wog, sW, 64, 64, tid);
  __syncthreads();

  // ---- Phase 6: output projection -> global, M=128 N=64 K=64 ---------------
  {
    float* og = out + (size_t)bidx * (LH * D_MODEL) + mb * 64 + ln15;
    v16bf a0 = ldsA(sH2 + (wave << 4) * 64 + 0,  64, lane);
    v16bf a1 = ldsA(sH2 + (wave << 4) * 64 + 32, 64, lane);
#pragma unroll
    for (int nt = 0; nt < 4; ++nt) {
      v8f acc = {};
      acc = WMMA_BF16(a0, ldsB(sW, 0 * 4 + nt, lane), acc);
      acc = WMMA_BF16(a1, ldsB(sW, 1 * 4 + nt, lane), acc);
      const float bias = bog[nt * 16 + ln15];
#pragma unroll
      for (int v = 0; v < 8; ++v)
        og[v * 64 + nt * 16] = acc[v] + bias;
    }
  }
}

extern "C" void kernel_launch(void* const* d_in, const int* in_sizes, int n_in,
                              void* d_out, int out_size, void* d_ws, size_t ws_size,
                              hipStream_t stream) {
  (void)n_in; (void)out_size; (void)d_ws; (void)ws_size;
  const float* hlain = (const float*)d_in[0];
  const float* pepin = (const float*)d_in[1];
  const float* Whla  = (const float*)d_in[2];
  const float* bhla  = (const float*)d_in[3];
  const float* Wpep  = (const float*)d_in[4];
  const float* bpep  = (const float*)d_in[5];
  const float* relb  = (const float*)d_in[6];
  const float* lng   = (const float*)d_in[7];
  const float* lnb   = (const float*)d_in[8];
  const float* W1    = (const float*)d_in[9];
  const float* b1    = (const float*)d_in[10];
  const float* W2    = (const float*)d_in[11];
  const float* b2    = (const float*)d_in[12];
  const float* Wout  = (const float*)d_in[13];
  const float* bout  = (const float*)d_in[14];
  float* out = (float*)d_out;

  const int B = in_sizes[0] / (LH * D_MODEL);   // 4096
  hla2pep_fused<<<B, 256, 0, stream>>>(hlain, pepin, Whla, bhla, Wpep, bpep, relb,
                                       lng, lnb, W1, b1, W2, b2, Wout, bout, out);
}
